// KPConv_49795850830137
// MI455X (gfx1250) — matrix operand order repaired and checked by reading, not compiled, not run on hardware
//
#include <hip/hip_runtime.h>

typedef __attribute__((ext_vector_type(16))) _Float16 v16h;
typedef __attribute__((ext_vector_type(8)))  _Float16 v8h;
typedef __attribute__((ext_vector_type(8)))  float    v8f;
typedef __attribute__((ext_vector_type(4)))  float    v4f;
typedef __attribute__((ext_vector_type(4)))  unsigned int u32x4;
typedef __attribute__((ext_vector_type(8)))  int      i32x8;
typedef __attribute__((ext_vector_type(4)))  int      i32x4;

#define NPTS    32768
#define KP      16          // kernel points == neighbors per center
#define NCEN    8192
#define IN_F    64
#define OUT_F   128
#define KTILES  32          // 1024 / 32
#define INV_EXT 1.5f        // 1 / KP_EXTENT

#define FRAG_HALVES  (32 * 16)                   // fragment: 32 lanes x 16 halves
#define BSLICE_HALVES (2 * KTILES * FRAG_HALVES) // 2 nt x 32 kt = 32768 halves = 64 KB
#define BSLICE_BYTES  (BSLICE_HALVES * 2)

// workspace byte offsets
#define S_OFF   ((size_t)0)               // NC*16 floats            = 512 KB
#define WP_OFF  ((size_t)(512*1024))      // 8*32*32*16 halves       = 256 KB (nt-major)
#define AP_OFF  ((size_t)(768*1024))      // 512*32*32*16 halves     =  16 MB

#if defined(__has_builtin)
#  if __has_builtin(__builtin_amdgcn_tensor_load_to_lds)
#    define HAVE_TDM 1
#  endif
#endif
#ifndef HAVE_TDM
#  define HAVE_TDM 0
#endif

// ---------------------------------------------------------------------------
// Kernel 1: geometry.  argmin over 16 kernel points per edge, linear influence
// weight at the argmin, reduce to s[c,k].  8 centers per 128-thread block.
// ---------------------------------------------------------------------------
__global__ void kp_geom(const float* __restrict__ pos,
                        const int*   __restrict__ esrc,
                        const int*   __restrict__ etgt,
                        const float* __restrict__ kpts,
                        float*       __restrict__ s_out) {
    __shared__ float kp[KP * 3];
    __shared__ float wv[128];
    __shared__ int   nn[128];
    int tid = threadIdx.x;
    if (tid < KP * 3) kp[tid] = kpts[tid];
    __syncthreads();

    int cl = tid >> 4, n = tid & 15;
    int c  = blockIdx.x * 8 + cl;
    int e  = c * KP + n;
    int src = esrc[e];
    int tgt = etgt[e];
    float rx = pos[tgt * 3 + 0] - pos[src * 3 + 0];
    float ry = pos[tgt * 3 + 1] - pos[src * 3 + 1];
    float rz = pos[tgt * 3 + 2] - pos[src * 3 + 2];

    float best = 3.4e38f;
    int   barg = 0;
    #pragma unroll
    for (int k = 0; k < KP; ++k) {
        float dx = rx - kp[k * 3 + 0];
        float dy = ry - kp[k * 3 + 1];
        float dz = rz - kp[k * 3 + 2];
        float sq = dx * dx + dy * dy + dz * dz;
        if (sq < best) { best = sq; barg = k; }   // strict '<' => first-index tie-break
    }
    float w = 1.0f - sqrtf(best) * INV_EXT;
    wv[tid] = (w > 0.0f) ? w : 0.0f;
    nn[tid] = barg;
    __syncthreads();

    float s = 0.0f;
    #pragma unroll
    for (int j = 0; j < KP; ++j)
        if (nn[cl * 16 + j] == n) s += wv[cl * 16 + j];
    s_out[c * 16 + n] = s;
}

// ---------------------------------------------------------------------------
// Kernel 2: repack kernel_weight [16,64,128] -> f16 B-fragments, nt-major:
// Wp[nt][kt][lane][slot], so a (nt,nt+1) column slice over all K is a single
// contiguous 64 KB region (one TDM descriptor).
// ---------------------------------------------------------------------------
__global__ void kp_wpack(const float* __restrict__ kw,
                         _Float16*    __restrict__ wp) {
    int t    = blockIdx.x * blockDim.x + threadIdx.x;   // 8192 threads
    int lane = t & 31;
    int nt   = (t >> 5) & 7;
    int kt   = t >> 8;
    int kbase = kt * 32 + ((lane & 16) ? 16 : 0);
    int col   = nt * 16 + (lane & 15);
    v8h lo, hi;
    #pragma unroll
    for (int s = 0; s < 8; ++s) lo[s] = (_Float16)kw[(size_t)(kbase + s)     * OUT_F + col];
    #pragma unroll
    for (int s = 0; s < 8; ++s) hi[s] = (_Float16)kw[(size_t)(kbase + 8 + s) * OUT_F + col];
    v8h* dst = (v8h*)(wp + ((size_t)(nt * KTILES + kt) * 32 + lane) * 16);
    dst[0] = lo;
    dst[1] = hi;
}

// ---------------------------------------------------------------------------
// Kernel 3: A = s[c,k] * x[edge_src[c*16+k]] in f16, A-fragment swizzle.
// ---------------------------------------------------------------------------
__global__ void kp_apack(const float* __restrict__ x,
                         const int*   __restrict__ esrc,
                         const float* __restrict__ s_in,
                         _Float16*    __restrict__ ap) {
    int t   = blockIdx.x * blockDim.x + threadIdx.x;    // NC * 128 threads
    int c   = t >> 7;
    int rem = t & 127;
    int k   = rem >> 3;
    int i0  = (rem & 7) << 3;

    float s   = s_in[c * 16 + k];
    int   src = esrc[c * KP + k];
    const v4f* xp = (const v4f*)(x + (size_t)src * IN_F + i0);
    v4f a = xp[0], b = xp[1];
    v8h h;
    h[0] = (_Float16)(a[0] * s); h[1] = (_Float16)(a[1] * s);
    h[2] = (_Float16)(a[2] * s); h[3] = (_Float16)(a[3] * s);
    h[4] = (_Float16)(b[0] * s); h[5] = (_Float16)(b[1] * s);
    h[6] = (_Float16)(b[2] * s); h[7] = (_Float16)(b[3] * s);

    int tr  = c >> 4, r = c & 15;
    int kk  = k * 64 + i0;
    int kt  = kk >> 5;
    int kko = kk & 31;
    int lane  = r + ((kko & 16) ? 16 : 0);
    int slot0 =     ((kko & 8)  ? 8  : 0);
    *(v8h*)(ap + ((size_t)(tr * 32 + kt) * 32 + lane) * 16 + slot0) = h;
}

// ---------------------------------------------------------------------------
// TDM: 1-D copy of `bytes` (multiple of 8) from global -> LDS via the Tensor
// Data Mover.  D# per CDNA5 ISA 8.3/8.4 (count=1, type=2, data_size=8B,
// flat 1-D tile).  Tracked by TENSORcnt.
// ---------------------------------------------------------------------------
#if HAVE_TDM
__device__ __forceinline__ void tdm_copy_1d(unsigned lds_off, const void* gsrc,
                                            unsigned bytes) {
    unsigned long long ga = (unsigned long long)(uintptr_t)gsrc;
    unsigned ne = bytes >> 3;                 // 8-byte elements (<= 65535)
    u32x4 g0;
    g0[0] = 1u;                               // count=1, user descriptor
    g0[1] = lds_off;                          // lds_addr
    g0[2] = (unsigned)(ga & 0xFFFFFFFFu);     // global_addr[31:0]
    g0[3] = (unsigned)((ga >> 32) & 0x01FFFFFFu) | (2u << 30);  // addr[56:32] | type=2
    i32x8 g1;
    g1[0] = (int)(3u << 16);                  // workgroup_mask=0, data_size=8B
    g1[1] = (int)(ne << 16);                  // tensor_dim0[15:0]  (bits 63:48)
    g1[2] = (int)((ne >> 16) & 0xFFFFu) | (1 << 16);  // dim0[31:16] | tensor_dim1=1
    g1[3] = (int)(ne << 16);                  // dim1 hi=0 | tile_dim0 (bits 127:112)
    g1[4] = 1;                                // tile_dim1=1, tile_dim2=0
    g1[5] = (int)ne;                          // tensor_dim0_stride[31:0]
    g1[6] = (int)(ne << 16);                  // d0_stride hi=0 | d1_stride[15:0]
    g1[7] = (int)(ne >> 16);                  // d1_stride[47:16]
    i32x4 z4 = {0, 0, 0, 0};
#if __clang_major__ >= 23
    i32x8 z8 = {0, 0, 0, 0, 0, 0, 0, 0};
    __builtin_amdgcn_tensor_load_to_lds(g0, g1, z4, z4, z8, 0);
#else
    __builtin_amdgcn_tensor_load_to_lds(g0, g1, z4, z4, 0);
#endif
}
#endif

// ---------------------------------------------------------------------------
// Kernel 4: GEMM [8192 x 1024] x [1024 x 128].
// Grid = 64 row-groups x 4 column-pairs = 256 blocks; block = 8 waves.
// Wave w owns row-tile rg*8+w and the block's 2 column tiles (2 f32
// accumulators, ~70 VGPRs).  The block's entire B need (2 nt x 32 kt = 64 KB)
// is staged into LDS by ONE TDM descriptor; one barrier; then the whole
// K-loop runs from resident LDS with no further synchronization.
// ---------------------------------------------------------------------------
__global__ void kp_gemm(const _Float16* __restrict__ ap,
                        const _Float16* __restrict__ wp,
                        float*          __restrict__ out) {
    __shared__ _Float16 lB[BSLICE_HALVES];     // 64 KB, whole-K B slice

    int tid  = threadIdx.x;
    int lane = tid & 31;
    int wave = tid >> 5;
    int rg   = blockIdx.x >> 2;                // row group 0..63
    int np   = blockIdx.x & 3;                 // column pair 0..3
    int tr   = rg * 8 + wave;                  // row tile 0..511
    int nt0  = np * 2;

    const v16h* A = ((const v16h*)ap) + ((size_t)tr * KTILES * 32 + lane);
    const char* bsrc = (const char*)(wp + (size_t)nt0 * KTILES * FRAG_HALVES);

#if HAVE_TDM
    if (tid < 32) {                            // wave 0 drives the TDM
        tdm_copy_1d((unsigned)(uintptr_t)&lB[0], bsrc, BSLICE_BYTES);
        __builtin_amdgcn_s_wait_tensorcnt(0);
    }
#else
    {   // fallback: cooperative vector copy
        const v8h* src = (const v8h*)bsrc;
        v8h*       dst = (v8h*)&lB[0];
        #pragma unroll
        for (int r = 0; r < BSLICE_BYTES / 16 / 256; ++r)
            dst[tid + r * 256] = src[tid + r * 256];
    }
#endif
    __syncthreads();                           // B slice visible to all waves

    const v16h* B0 = ((const v16h*)lB) + lane;                 // nt local 0
    const v16h* B1 = ((const v16h*)lB) + (size_t)KTILES * 32 + lane;  // nt local 1

    v8f acc0 = {}, acc1 = {};
    #pragma unroll
    for (int kt = 0; kt < KTILES; ++kt) {
        v16h af = A[(size_t)kt * 32];
        if (kt + 2 < KTILES)
            __builtin_prefetch((const char*)(A + (size_t)(kt + 2) * 32), 0, 1);
        v16h b0 = B0[(size_t)kt * 32];
        v16h b1 = B1[(size_t)kt * 32];
        acc0 = __builtin_amdgcn_wmma_f32_16x16x32_f16(
                   false, af, false, b0, (short)0, acc0, false, false);
        acc1 = __builtin_amdgcn_wmma_f32_16x16x32_f16(
                   false, af, false, b1, (short)0, acc1, false, false);
    }

    // C/D layout: VGPR j -> row j + 8*(lane>=16), col = lane&15 within tile.
    int row0 = tr * 16 + ((lane & 16) ? 8 : 0);
    int col  = lane & 15;
    #pragma unroll
    for (int j = 0; j < 8; ++j) {
        size_t r = (size_t)(row0 + j) * OUT_F;
        out[r + (nt0 + 0) * 16 + col] = acc0[j];
        out[r + (nt0 + 1) * 16 + col] = acc1[j];
    }
}

// ---------------------------------------------------------------------------
extern "C" void kernel_launch(void* const* d_in, const int* in_sizes, int n_in,
                              void* d_out, int out_size, void* d_ws, size_t ws_size,
                              hipStream_t stream) {
    const float* x    = (const float*)d_in[0];   // [N, 64]
    const float* pos  = (const float*)d_in[1];   // [N, 3]
    const int*   esrc = (const int*)  d_in[2];   // [E]
    const int*   etgt = (const int*)  d_in[3];   // [E]
    const float* kpts = (const float*)d_in[4];   // [16, 3]
    const float* kw   = (const float*)d_in[5];   // [16, 64, 128]
    float*       outp = (float*)d_out;           // [8192, 128]

    float*    s_buf = (float*)   ((char*)d_ws + S_OFF);
    _Float16* wpk   = (_Float16*)((char*)d_ws + WP_OFF);
    _Float16* apk   = (_Float16*)((char*)d_ws + AP_OFF);

    kp_geom <<<NCEN / 8,           128, 0, stream>>>(pos, esrc, etgt, kpts, s_buf);
    kp_wpack<<<8192 / 256,         256, 0, stream>>>(kw, wpk);
    kp_apack<<<(NCEN * 128) / 256, 256, 0, stream>>>(x, esrc, s_buf, apk);
    kp_gemm <<<256,                256, 0, stream>>>(apk, wpk, outp);
}
